// LongformerMaxp_68753836474369
// MI455X (gfx1250) — matrix operand order repaired and checked by reading, not compiled, not run on hardware
//
#include <hip/hip_runtime.h>
#include <math.h>

typedef __attribute__((ext_vector_type(16))) __bf16 v16bf;
typedef __attribute__((ext_vector_type(8)))  __bf16 bf16x8;
typedef __attribute__((ext_vector_type(8)))  float   v8f;
typedef __attribute__((ext_vector_type(4)))  unsigned int u32x4;
typedef __attribute__((ext_vector_type(8)))  int i32x8;
typedef __attribute__((ext_vector_type(4)))  int i32x4;

#define HDIM   768
#define SEQN   1024
#define NHEAD  12
#define DHEAD  64
#define FFDIM  3072
#define NLAY   12
#define NBATCH 6
#define NTOK   (NBATCH*SEQN)
#define WWIN   256
#define GGLB   128
#define NEGV   (-1.0e9f)
#define QSCALE 0.125f
#define LNEPS  1e-5f
#define LDA    40   // LDS row stride (bf16 elems): 32 data + 4-DWORD pad = 80B

#if defined(__has_builtin)
#if __has_builtin(__builtin_amdgcn_tensor_load_to_lds) && \
    __has_builtin(__builtin_amdgcn_s_wait_tensorcnt)
#define USE_TDM 1
#endif
#endif
#ifndef USE_TDM
#define USE_TDM 0
#endif

// ---------------------------------------------------------------- helpers ----

static __device__ inline v8f wmma_bf16(v16bf a, v16bf b, v8f c) {
  return __builtin_amdgcn_wmma_f32_16x16x32_bf16(false, a, false, b, (short)0, c,
                                                 false, false);
}

// A-fragment (16x32 bf16): lane row = lane&15; half = (lane>>4)*8.
// Caller passes p = rowPtr + half; reads 8 elems at p and 8 at p+16.
static __device__ inline v16bf pack_a(const __bf16* p) {
  bf16x8 lo = *(const bf16x8*)(p);
  bf16x8 hi = *(const bf16x8*)(p + 16);
  v16bf f;
#pragma unroll
  for (int i = 0; i < 8; i++) { f[i] = lo[i]; f[i + 8] = hi[i]; }
  return f;
}

// B-fragment (32x16 bf16): lane col = lane&15; half = (lane>>4)*8.
// Caller passes p = colRowPtr + half; reads 8 elems at p and 8 at p+16.
static __device__ inline v16bf pack_b(const __bf16* p) {
  bf16x8 lo = *(const bf16x8*)(p);
  bf16x8 hi = *(const bf16x8*)(p + 16);
  v16bf f;
#pragma unroll
  for (int i = 0; i < 8; i++) { f[i] = lo[i]; f[i + 8] = hi[i]; }
  return f;
}

static __device__ inline float gelu_f(float x) {
  return 0.5f * x * (1.0f + tanhf(0.7978845608028654f * (x + 0.044715f * x * x * x)));
}

static __device__ inline int iclamp(int v, int lo, int hi) {
  return v < lo ? lo : (v > hi ? hi : v);
}

#if USE_TDM
// Issue a 2D TDM tile load: tile_d1 lines of tile_d0 bf16 elements, line
// stride = stride_d0 elements, landing packed in LDS with 16B pad per 64B
// (-> LDA=40 rows). D# packing per CDNA5 ISA.
static __device__ inline void tdm_load_tile_2d(unsigned lds,
                                               const __bf16* gptr,
                                               unsigned tensor_d0,
                                               unsigned tensor_d1,
                                               unsigned tile_d0,
                                               unsigned tile_d1,
                                               unsigned long long stride_d0) {
  unsigned long long ga = (unsigned long long)(size_t)gptr;
  u32x4 g0;
  g0[0] = 1u;                                      // count=1, user descriptor
  g0[1] = lds;                                     // lds_addr (bytes)
  g0[2] = (unsigned)(ga & 0xFFFFFFFFu);            // global_addr[31:0]
  g0[3] = (unsigned)((ga >> 32) & 0x01FFFFFFu)     // global_addr[56:32]
          | (2u << 30);                            // type = 2 ("image")
  i32x8 g1;
  unsigned w0 = (1u << 16)                         // data_size = 2 bytes
              | (1u << 20)                         // pad_enable
              | (3u << 22)                         // pad_interval: 16 DWORDs
              | (3u << 25);                        // pad_amount: 4 DWORDs
  g1[0] = (int)w0;
  g1[1] = (int)((tensor_d0 & 0xFFFFu) << 16);                    // dim0[15:0]
  g1[2] = (int)((tensor_d0 >> 16) | ((tensor_d1 & 0xFFFFu) << 16));
  g1[3] = (int)((tensor_d1 >> 16) | (tile_d0 << 16));            // tile_dim0
  g1[4] = (int)(tile_d1 & 0xFFFFu);                // tile_dim1; tile_dim2=0
  g1[5] = (int)(stride_d0 & 0xFFFFFFFFull);        // dim0_stride[31:0]
  g1[6] = (int)((stride_d0 >> 32) & 0xFFFFull);    // dim0_stride[47:32]
  g1[7] = 0;
  i32x4 z4 = {0, 0, 0, 0};
  i32x8 z8 = {0, 0, 0, 0, 0, 0, 0, 0};
  __builtin_amdgcn_tensor_load_to_lds(g0, g1, z4, z4, z8, 0);
}
static __device__ inline unsigned lds_addr_of(const void* p) {
  return (unsigned)(size_t)p;
}
#endif

// ------------------------------------------------------------ fp32 -> bf16 ---

__global__ __launch_bounds__(256) void cvt_f2b_kernel(const float* __restrict__ s,
                                                      __bf16* __restrict__ d, int n) {
  int i = blockIdx.x * 256 + threadIdx.x;
  if (i < n) d[i] = (__bf16)s[i];
}

// fp32 (NLAY stacked KxN) -> bf16 transposed (NLAY stacked NxK)
__global__ __launch_bounds__(256) void cvt_t_kernel(const float* __restrict__ s,
                                                    __bf16* __restrict__ d,
                                                    int K, int N) {
  size_t per = (size_t)K * N;
  size_t e = (size_t)blockIdx.x * 256 + threadIdx.x;
  if (e >= (size_t)NLAY * per) return;
  size_t l = e / per, r = e - l * per;
  int k = (int)(r / (size_t)N), n = (int)(r - (size_t)k * N);
  d[l * per + (size_t)n * K + k] = (__bf16)s[e];
}

// ------------------------------------------------------------------- GEMM ----
// C(MxN) = A(MxK,bf16) @ Bt^T + bias, where Bt is the N-major (NxK) weight.
// Block tile 128x128, 8 waves of 64x32, K-steps of 32.
// TDM path: double-buffered LDS tiles DMA'd by the Tensor Data Mover.

__global__ __launch_bounds__(256) void gemm_bf16_kernel(
    const __bf16* __restrict__ A, const __bf16* __restrict__ Bt,
    const float* __restrict__ bias, float* __restrict__ Cf,
    __bf16* __restrict__ Cb, int M, int N, int K, int act, float outScale) {
#if USE_TDM
  __shared__ __align__(16) __bf16 As[2][128 * LDA];
  __shared__ __align__(16) __bf16 Bs[2][128 * LDA];
#else
  __shared__ __align__(16) __bf16 As1[128 * LDA];
  __shared__ __align__(16) __bf16 Bs1[128 * LDA];
#endif
  int tid = threadIdx.x, lane = tid & 31, wave = tid >> 5;
  int m0 = blockIdx.y * 128, n0 = blockIdx.x * 128;
  int wm = (wave & 1) * 64, wn = (wave >> 1) * 32;
  int rl = lane & 15, lh = lane >> 4;
  int half = lh * 8, khs = lh * 16;

  v8f zero = {0.f, 0.f, 0.f, 0.f, 0.f, 0.f, 0.f, 0.f};
  v8f acc[4][2];
#pragma unroll
  for (int i = 0; i < 4; i++)
#pragma unroll
    for (int j = 0; j < 2; j++) acc[i][j] = zero;

#if USE_TDM
  int cur = 0;
  if (tid < 32) {  // wave 0 drives the TDM (uniform branch; EXEC-independent op)
    tdm_load_tile_2d(lds_addr_of(&As[0][0]), A + (size_t)m0 * K, (unsigned)K,
                     (unsigned)M, 32u, 128u, (unsigned long long)K);
    tdm_load_tile_2d(lds_addr_of(&Bs[0][0]), Bt + (size_t)n0 * K, (unsigned)K,
                     (unsigned)N, 32u, 128u, (unsigned long long)K);
  }
  for (int k0 = 0; k0 < K; k0 += 32) {
    if (tid < 32) __builtin_amdgcn_s_wait_tensorcnt(0);
    __syncthreads();  // tiles[cur] ready for all waves
    if (k0 + 32 < K && tid < 32) {
      tdm_load_tile_2d(lds_addr_of(&As[cur ^ 1][0]),
                       A + (size_t)m0 * K + (k0 + 32), (unsigned)K, (unsigned)M,
                       32u, 128u, (unsigned long long)K);
      tdm_load_tile_2d(lds_addr_of(&Bs[cur ^ 1][0]),
                       Bt + (size_t)n0 * K + (k0 + 32), (unsigned)K, (unsigned)N,
                       32u, 128u, (unsigned long long)K);
    }
    const __bf16* Ab = &As[cur][0];
    const __bf16* Bb = &Bs[cur][0];
#else
  for (int k0 = 0; k0 < K; k0 += 32) {
    // stage A tile 128x32 and Bt tile 128x32 (both K-contiguous): b128 only
#pragma unroll
    for (int i = 0; i < 2; i++) {
      int idx = tid + i * 256;
      int r = idx >> 2, c = (idx & 3) * 8;
      *(bf16x8*)&As1[r * LDA + c] =
          *(const bf16x8*)(A + (size_t)(m0 + r) * K + k0 + c);
      *(bf16x8*)&Bs1[r * LDA + c] =
          *(const bf16x8*)(Bt + (size_t)(n0 + r) * K + k0 + c);
    }
    if (k0 + 32 < K) {  // prefetch next K tile
      __builtin_prefetch(A + (size_t)(m0 + (tid >> 1)) * K + k0 + 32 + (tid & 1) * 16, 0, 1);
      __builtin_prefetch(Bt + (size_t)(n0 + (tid >> 1)) * K + k0 + 32 + (tid & 1) * 16, 0, 1);
    }
    __syncthreads();
    const __bf16* Ab = As1;
    const __bf16* Bb = Bs1;
#endif
    v16bf bfr[2];
#pragma unroll
    for (int ni = 0; ni < 2; ni++)
      bfr[ni] = pack_b(&Bb[(wn + ni * 16 + rl) * LDA + half]);
#pragma unroll
    for (int mi = 0; mi < 4; mi++) {
      v16bf af = pack_a(&Ab[(wm + mi * 16 + rl) * LDA + half]);
#pragma unroll
      for (int ni = 0; ni < 2; ni++)
        acc[mi][ni] = wmma_bf16(af, bfr[ni], acc[mi][ni]);
    }
    __syncthreads();
#if USE_TDM
    cur ^= 1;
#endif
  }

  // epilogue: C layout -> m = r + (lane>=16 ? 8 : 0), n = lane&15
#pragma unroll
  for (int mi = 0; mi < 4; mi++) {
#pragma unroll
    for (int ni = 0; ni < 2; ni++) {
      int gn = n0 + wn + ni * 16 + rl;
      float bvv = bias ? bias[gn] : 0.f;
#pragma unroll
      for (int r = 0; r < 8; r++) {
        int gm = m0 + wm + mi * 16 + r + lh * 8;
        float v = acc[mi][ni][r] + bvv;
        if (act == 1) v = gelu_f(v);
        v *= outScale;
        if (Cf) Cf[(size_t)gm * N + gn] = v;
        if (Cb) Cb[(size_t)gm * N + gn] = (__bf16)v;
      }
    }
  }
}

// ------------------------------------------------- embedding + layer norm ----

__global__ __launch_bounds__(256) void embed_ln_kernel(
    const int* __restrict__ ids, const float* __restrict__ ew,
    const float* __restrict__ ep, const float* __restrict__ g,
    const float* __restrict__ be, float* __restrict__ xo,
    __bf16* __restrict__ xb) {
  int t = blockIdx.x;
  int s = t & (SEQN - 1);
  int id = ids[t];
  int tid = threadIdx.x, lane = tid & 31, wave = tid >> 5;
  __shared__ float rbuf[16];
  float vals[3], sum = 0.f, sq = 0.f;
#pragma unroll
  for (int i = 0; i < 3; i++) {
    int c = tid + i * 256;
    float v = ew[(size_t)id * HDIM + c] + ep[(size_t)s * HDIM + c];
    vals[i] = v; sum += v; sq += v * v;
  }
#pragma unroll
  for (int o = 16; o; o >>= 1) { sum += __shfl_xor(sum, o, 32); sq += __shfl_xor(sq, o, 32); }
  if (lane == 0) { rbuf[wave] = sum; rbuf[8 + wave] = sq; }
  __syncthreads();
  if (tid == 0) {
    float S = 0, Q = 0;
    for (int i = 0; i < 8; i++) { S += rbuf[i]; Q += rbuf[8 + i]; }
    rbuf[0] = S; rbuf[8] = Q;
  }
  __syncthreads();
  float mean = rbuf[0] * (1.f / HDIM);
  float var = rbuf[8] * (1.f / HDIM) - mean * mean;
  float inv = rsqrtf(var + LNEPS);
#pragma unroll
  for (int i = 0; i < 3; i++) {
    int c = tid + i * 256;
    float v = (vals[i] - mean) * inv * g[c] + be[c];
    xo[(size_t)t * HDIM + c] = v;
    xb[(size_t)t * HDIM + c] = (__bf16)v;
  }
}

// ------------------------------------------------- residual + layer norm -----

__global__ __launch_bounds__(256) void add_ln_kernel(
    const float* __restrict__ x, const float* __restrict__ a,
    const float* __restrict__ g, const float* __restrict__ be,
    float* __restrict__ xo, __bf16* __restrict__ xb) {
  int t = blockIdx.x;
  int tid = threadIdx.x, lane = tid & 31, wave = tid >> 5;
  __shared__ float rbuf[16];
  float vals[3], sum = 0.f, sq = 0.f;
#pragma unroll
  for (int i = 0; i < 3; i++) {
    int c = tid + i * 256;
    float v = x[(size_t)t * HDIM + c] + a[(size_t)t * HDIM + c];
    vals[i] = v; sum += v; sq += v * v;
  }
#pragma unroll
  for (int o = 16; o; o >>= 1) { sum += __shfl_xor(sum, o, 32); sq += __shfl_xor(sq, o, 32); }
  if (lane == 0) { rbuf[wave] = sum; rbuf[8 + wave] = sq; }
  __syncthreads();
  if (tid == 0) {
    float S = 0, Q = 0;
    for (int i = 0; i < 8; i++) { S += rbuf[i]; Q += rbuf[8 + i]; }
    rbuf[0] = S; rbuf[8] = Q;
  }
  __syncthreads();
  float mean = rbuf[0] * (1.f / HDIM);
  float var = rbuf[8] * (1.f / HDIM) - mean * mean;
  float inv = rsqrtf(var + LNEPS);
#pragma unroll
  for (int i = 0; i < 3; i++) {
    int c = tid + i * 256;
    float v = (vals[i] - mean) * inv * g[c] + be[c];
    xo[(size_t)t * HDIM + c] = v;
    xb[(size_t)t * HDIM + c] = (__bf16)v;
  }
}

// ------------------------------------------ local (banded+global) attention --
// grid: (16 query-chunks, 4 window-blocks, NBATCH*NHEAD). 16 queries/chunk.
// columns: [0,128) global keys (kg/vg), [128,896) banded local keys (k/v).

__global__ __launch_bounds__(256) void attn_local_kernel(
    const __bf16* __restrict__ q, const __bf16* __restrict__ kk,
    const __bf16* __restrict__ vv, const __bf16* __restrict__ kg,
    const __bf16* __restrict__ vg, const float* __restrict__ mask,
    float* __restrict__ ctx) {
  __shared__ float sc[16 * 896];
  int tid = threadIdx.x, lane = tid & 31, wave = tid >> 5;
  int chunk = blockIdx.x, nblk = blockIdx.y, bh = blockIdx.z;
  int b = bh / NHEAD, h = bh % NHEAD;
  int q0 = nblk * WWIN + chunk * 16;
  const size_t base = (size_t)b * SEQN;
  int rl = lane & 15, lh = lane >> 4;
  int half = lh * 8, khs = lh * 16;

  const __bf16* qrow = q + (base + q0 + rl) * HDIM + h * DHEAD;
  v16bf afrag[2];
#pragma unroll
  for (int ks = 0; ks < 2; ks++) afrag[ks] = pack_a(qrow + ks * 32 + half);

  v8f zero = {0.f, 0.f, 0.f, 0.f, 0.f, 0.f, 0.f, 0.f};

  for (int t = wave; t < 56; t += 8) {
    int col = t * 16 + rl;
    const __bf16* krow;
    if (col < GGLB) {
      krow = kg + (base + col) * HDIM + h * DHEAD;
    } else {
      int kp = nblk * WWIN + (col - GGLB) - WWIN;
      krow = kk + (base + iclamp(kp, 0, SEQN - 1)) * HDIM + h * DHEAD;
    }
    v8f accs = zero;
#pragma unroll
    for (int ks = 0; ks < 2; ks++)
      accs = wmma_bf16(afrag[ks], pack_b(krow + ks * 32 + half), accs);

    int c = col;
    bool mkcol;
    if (c < GGLB) {
      mkcol = mask[base + c] > 0.5f;
    } else {
      int kp = nblk * WWIN + (c - GGLB) - WWIN;
      mkcol = (kp >= GGLB) && (kp < SEQN) && (mask[base + kp] > 0.5f);
    }
#pragma unroll
    for (int r = 0; r < 8; r++) {
      int m = r + lh * 8;
      bool ok = mkcol;
      if (c >= GGLB) {
        int off = (c - GGLB) - (WWIN + chunk * 16 + m);
        ok = ok && (off >= -WWIN) && (off <= WWIN);
      }
      sc[m * 896 + c] = ok ? accs[r] : NEGV;
    }
  }
  __syncthreads();

  {
    int row = tid >> 4, j = tid & 15;
    float mx = -3.0e38f;
    for (int c = j; c < 896; c += 16) mx = fmaxf(mx, sc[row * 896 + c]);
#pragma unroll
    for (int o = 8; o; o >>= 1) mx = fmaxf(mx, __shfl_xor(mx, o, 16));
    float sum = 0.f;
    for (int c = j; c < 896; c += 16) {
      float e = __expf(sc[row * 896 + c] - mx);
      sc[row * 896 + c] = e; sum += e;
    }
#pragma unroll
    for (int o = 8; o; o >>= 1) sum += __shfl_xor(sum, o, 16);
    float inv = 1.0f / sum;
    for (int c = j; c < 896; c += 16) sc[row * 896 + c] *= inv;
  }
  __syncthreads();

  int nt = wave & 3, khf = wave >> 2;
  v8f accc = zero;
  for (int ks = 0; ks < 14; ks++) {
    int cb = khf * 448 + ks * 32;
    v16bf af;
#pragma unroll
    for (int i = 0; i < 8; i++) {
      af[i]     = (__bf16)sc[rl * 896 + cb + half + i];
      af[i + 8] = (__bf16)sc[rl * 896 + cb + half + 16 + i];
    }
    v16bf bf_;
    int n = nt * 16 + rl;
#pragma unroll
    for (int i = 0; i < 16; i++) {
      int c = cb + half + i + (i & 8);
      const __bf16* vrow;
      if (c < GGLB) {
        vrow = vg + (base + c) * HDIM + h * DHEAD;
      } else {
        int kp = nblk * WWIN + (c - GGLB) - WWIN;
        vrow = vv + (base + iclamp(kp, 0, SEQN - 1)) * HDIM + h * DHEAD;
      }
      bf_[i] = vrow[n];
    }
    accc = wmma_bf16(af, bf_, accc);
  }
  __syncthreads();
#pragma unroll
  for (int r = 0; r < 8; r++) {
    int m = r + lh * 8;
    sc[wave * 256 + m * 16 + rl] = accc[r];
  }
  __syncthreads();
  for (int i = tid; i < 1024; i += 256) {
    int nt2 = i >> 8, rem = i & 255, m = rem >> 4, d = rem & 15;
    float val = sc[nt2 * 256 + rem] + sc[(nt2 + 4) * 256 + rem];
    ctx[(base + q0 + m) * HDIM + h * DHEAD + nt2 * 16 + d] = val;
  }
}

// ----------------------------------------------- global-row full attention ---

__global__ __launch_bounds__(256) void attn_global_kernel(
    const __bf16* __restrict__ qg, const __bf16* __restrict__ kg,
    const __bf16* __restrict__ vg, const float* __restrict__ mask,
    float* __restrict__ ctx) {
  __shared__ float sc[16 * 1024];
  int tid = threadIdx.x, lane = tid & 31, wave = tid >> 5;
  int chunk = blockIdx.x, bh = blockIdx.z;
  int b = bh / NHEAD, h = bh % NHEAD;
  int q0 = chunk * 16;
  const size_t base = (size_t)b * SEQN;
  int rl = lane & 15, lh = lane >> 4;
  int half = lh * 8, khs = lh * 16;

  const __bf16* qrow = qg + (base + q0 + rl) * HDIM + h * DHEAD;
  v16bf afrag[2];
#pragma unroll
  for (int ks = 0; ks < 2; ks++) afrag[ks] = pack_a(qrow + ks * 32 + half);

  v8f zero = {0.f, 0.f, 0.f, 0.f, 0.f, 0.f, 0.f, 0.f};

  for (int t = wave; t < 64; t += 8) {
    int col = t * 16 + rl;
    const __bf16* krow = kg + (base + col) * HDIM + h * DHEAD;
    v8f accs = zero;
#pragma unroll
    for (int ks = 0; ks < 2; ks++)
      accs = wmma_bf16(afrag[ks], pack_b(krow + ks * 32 + half), accs);
    bool mkcol = mask[base + col] > 0.5f;
#pragma unroll
    for (int r = 0; r < 8; r++) {
      int m = r + lh * 8;
      sc[m * 1024 + col] = mkcol ? accs[r] : NEGV;
    }
  }
  __syncthreads();
  {
    int row = tid >> 4, j = tid & 15;
    float mx = -3.0e38f;
    for (int c = j; c < 1024; c += 16) mx = fmaxf(mx, sc[row * 1024 + c]);
#pragma unroll
    for (int o = 8; o; o >>= 1) mx = fmaxf(mx, __shfl_xor(mx, o, 16));
    float sum = 0.f;
    for (int c = j; c < 1024; c += 16) {
      float e = __expf(sc[row * 1024 + c] - mx);
      sc[row * 1024 + c] = e; sum += e;
    }
#pragma unroll
    for (int o = 8; o; o >>= 1) sum += __shfl_xor(sum, o, 16);
    float inv = 1.0f / sum;
    for (int c = j; c < 1024; c += 16) sc[row * 1024 + c] *= inv;
  }
  __syncthreads();

  int nt = wave & 3, khf = wave >> 2;
  v8f accc = zero;
  for (int ks = 0; ks < 16; ks++) {
    int cb = khf * 512 + ks * 32;
    v16bf af;
#pragma unroll
    for (int i = 0; i < 8; i++) {
      af[i]     = (__bf16)sc[rl * 1024 + cb + half + i];
      af[i + 8] = (__bf16)sc[rl * 1024 + cb + half + 16 + i];
    }
    v16bf bf_;
    int n = nt * 16 + rl;
#pragma unroll
    for (int i = 0; i < 16; i++) {
      int c = cb + half + i + (i & 8);
      bf_[i] = vg[(base + c) * HDIM + h * DHEAD + n];
    }
    accc = wmma_bf16(af, bf_, accc);
  }
  __syncthreads();
#pragma unroll
  for (int r = 0; r < 8; r++) {
    int m = r + lh * 8;
    sc[wave * 256 + m * 16 + rl] = accc[r];
  }
  __syncthreads();
  for (int i = tid; i < 1024; i += 256) {
    int nt2 = i >> 8, rem = i & 255, m = rem >> 4, d = rem & 15;
    float val = sc[nt2 * 256 + rem] + sc[(nt2 + 4) * 256 + rem];
    ctx[(base + q0 + m) * HDIM + h * DHEAD + nt2 * 16 + d] = val;  // rows < G
  }
}

// ------------------------------------------------------------- classifier ----

__global__ __launch_bounds__(256) void head_kernel(
    const float* __restrict__ x, const float* __restrict__ Wd,
    const float* __restrict__ bd, const float* __restrict__ Wh,
    const float* __restrict__ bh, float* __restrict__ out) {
  __shared__ float cls[2 * HDIM];
  __shared__ float dbuf[2 * 128];
  int tid = threadIdx.x;
  for (int i = tid; i < 2 * HDIM; i += 256) {
    int bb = i / HDIM, c = i % HDIM;
    float m = x[(size_t)((bb * 3 + 0) * SEQN) * HDIM + c];
    m = fmaxf(m, x[(size_t)((bb * 3 + 1) * SEQN) * HDIM + c]);
    m = fmaxf(m, x[(size_t)((bb * 3 + 2) * SEQN) * HDIM + c]);
    cls[i] = m;
  }
  __syncthreads();
  for (int i = tid; i < 2 * 128; i += 256) {
    int bb = i >> 7, n = i & 127;
    float s = bd[n];
    for (int c = 0; c < HDIM; c++) s += cls[bb * HDIM + c] * Wd[c * 128 + n];
    dbuf[i] = tanhf(s);
  }
  __syncthreads();
  if (tid < 4) {
    int bb = tid >> 1, o = tid & 1;
    float s = bh[o];
    for (int c = 0; c < 128; c++) s += dbuf[bb * 128 + c] * Wh[c * 2 + o];
    out[bb * 2 + o] = s;      // score
    out[4 + bb * 2 + o] = s;  // logits (identical values)
  }
}

// -------------------------------------------------------------------- host ---

extern "C" void kernel_launch(void* const* d_in, const int* in_sizes, int n_in,
                              void* d_out, int out_size, void* d_ws,
                              size_t ws_size, hipStream_t stream) {
  (void)in_sizes; (void)out_size;
  if (n_in < 33) return;
  const int*   ids      = (const int*)d_in[0];
  const float* maskf    = (const float*)d_in[1];
  const float* emb_word = (const float*)d_in[3];
  const float* emb_pos  = (const float*)d_in[4];
  const float* ln_e_g   = (const float*)d_in[5];
  const float* ln_e_b   = (const float*)d_in[6];
  const float* Wq  = (const float*)d_in[7];  const float* bq  = (const float*)d_in[8];
  const float* Wk  = (const float*)d_in[9];  const float* bk  = (const float*)d_in[10];
  const float* Wv  = (const float*)d_in[11]; const float* bv  = (const float*)d_in[12];
  const float* Wo  = (const float*)d_in[13]; const float* bo  = (const float*)d_in[14];
  const float* Wqg = (const float*)d_in[15]; const float* bqg = (const float*)d_in[16];
  const float* Wkg = (const float*)d_in[17]; const float* bkg = (const float*)d_in[18];
  const float* Wvg = (const float*)d_in[19]; const float* bvg = (const float*)d_in[20];
  const float* ln1g = (const float*)d_in[21]; const float* ln1b = (const float*)d_in[22];
  const float* W1  = (const float*)d_in[23]; const float* b1  = (const float*)d_in[24];
  const float* W2  = (const float*)d_in[25]; const float* b2  = (const float*)d_in[26];
  const float* ln2g = (const float*)d_in[27]; const float* ln2b = (const float*)d_in[28];
  const float* Wd  = (const float*)d_in[29]; const float* bd  = (const float*)d_in[30];
  const float* Wh  = (const float*)d_in[31]; const float* bhh = (const float*)d_in[32];
  float* out = (float*)d_out;

  const size_t E  = (size_t)HDIM * HDIM;
  const size_t E1 = (size_t)HDIM * FFDIM;
  size_t off = 0;
  auto alloc = [&](size_t bytes) -> void* {
    off = (off + 255) & ~(size_t)255;
    void* p = (char*)d_ws + off;
    off += bytes;
    return p;
  };
  __bf16* wq_b  = (__bf16*)alloc(NLAY * E * 2);
  __bf16* wk_b  = (__bf16*)alloc(NLAY * E * 2);
  __bf16* wv_b  = (__bf16*)alloc(NLAY * E * 2);
  __bf16* wo_b  = (__bf16*)alloc(NLAY * E * 2);
  __bf16* wqg_b = (__bf16*)alloc(NLAY * E * 2);
  __bf16* wkg_b = (__bf16*)alloc(NLAY * E * 2);
  __bf16* wvg_b = (__bf16*)alloc(NLAY * E * 2);
  __bf16* w1_b  = (__bf16*)alloc(NLAY * E1 * 2);
  __bf16* w2_b  = (__bf16*)alloc(NLAY * E1 * 2);
  float*  xf   = (float*)alloc((size_t)NTOK * HDIM * 4);
  __bf16* xb   = (__bf16*)alloc((size_t)NTOK * HDIM * 2);
  __bf16* qb   = (__bf16*)alloc((size_t)NTOK * HDIM * 2);
  __bf16* kb   = (__bf16*)alloc((size_t)NTOK * HDIM * 2);
  __bf16* vb   = (__bf16*)alloc((size_t)NTOK * HDIM * 2);
  __bf16* kgb  = (__bf16*)alloc((size_t)NTOK * HDIM * 2);
  __bf16* vgb  = (__bf16*)alloc((size_t)NTOK * HDIM * 2);
  __bf16* qgb  = (__bf16*)alloc((size_t)NTOK * HDIM * 2);
  float*  ctx  = (float*)alloc((size_t)NTOK * HDIM * 4);
  __bf16* ctxb = (__bf16*)alloc((size_t)NTOK * HDIM * 2);
  float*  attn = (float*)alloc((size_t)NTOK * HDIM * 4);
  __bf16* f1b  = (__bf16*)alloc((size_t)NTOK * FFDIM * 2);
  float*  f2   = (float*)alloc((size_t)NTOK * HDIM * 4);
  if (off > ws_size) return;

  auto cvt = [&](const float* s, __bf16* d, size_t n) {
    cvt_f2b_kernel<<<dim3((unsigned)((n + 255) / 256)), 256, 0, stream>>>(s, d, (int)n);
  };
  auto cvtT = [&](const float* s, __bf16* d, int K, int N) {
    size_t tot = (size_t)NLAY * K * N;
    cvt_t_kernel<<<dim3((unsigned)((tot + 255) / 256)), 256, 0, stream>>>(s, d, K, N);
  };
  auto gemm = [&](const __bf16* A, const __bf16* Bm, const float* bias, float* Cf,
                  __bf16* Cb, int M, int N, int K, int act, float sc) {
    gemm_bf16_kernel<<<dim3(N / 128, M / 128), 256, 0, stream>>>(A, Bm, bias, Cf,
                                                                 Cb, M, N, K, act, sc);
  };

  // weights: fp32 -> bf16, transposed to N-major (out-dim major) once per call
  cvtT(Wq,  wq_b,  HDIM, HDIM);  cvtT(Wk,  wk_b,  HDIM, HDIM);
  cvtT(Wv,  wv_b,  HDIM, HDIM);  cvtT(Wo,  wo_b,  HDIM, HDIM);
  cvtT(Wqg, wqg_b, HDIM, HDIM);  cvtT(Wkg, wkg_b, HDIM, HDIM);
  cvtT(Wvg, wvg_b, HDIM, HDIM);
  cvtT(W1,  w1_b,  HDIM, FFDIM); cvtT(W2,  w2_b,  FFDIM, HDIM);

  embed_ln_kernel<<<NTOK, 256, 0, stream>>>(ids, emb_word, emb_pos, ln_e_g,
                                            ln_e_b, xf, xb);

  for (int l = 0; l < NLAY; l++) {
    const __bf16* wq = wq_b + (size_t)l * E;   const float* bql = bq + l * HDIM;
    const __bf16* wk = wk_b + (size_t)l * E;   const float* bkl = bk + l * HDIM;
    const __bf16* wv = wv_b + (size_t)l * E;   const float* bvl = bv + l * HDIM;
    const __bf16* wo = wo_b + (size_t)l * E;   const float* bol = bo + l * HDIM;
    const __bf16* wqg = wqg_b + (size_t)l * E; const float* bqgl = bqg + l * HDIM;
    const __bf16* wkg = wkg_b + (size_t)l * E; const float* bkgl = bkg + l * HDIM;
    const __bf16* wvg = wvg_b + (size_t)l * E; const float* bvgl = bvg + l * HDIM;
    const __bf16* w1 = w1_b + (size_t)l * E1;  const float* b1l = b1 + l * FFDIM;
    const __bf16* w2 = w2_b + (size_t)l * E1;  const float* b2l = b2 + l * HDIM;

    gemm(xb, wq,  bql,  nullptr, qb,  NTOK, HDIM, HDIM, 0, QSCALE);
    gemm(xb, wk,  bkl,  nullptr, kb,  NTOK, HDIM, HDIM, 0, 1.0f);
    gemm(xb, wv,  bvl,  nullptr, vb,  NTOK, HDIM, HDIM, 0, 1.0f);
    gemm(xb, wkg, bkgl, nullptr, kgb, NTOK, HDIM, HDIM, 0, 1.0f);
    gemm(xb, wvg, bvgl, nullptr, vgb, NTOK, HDIM, HDIM, 0, 1.0f);
    gemm(xb, wqg, bqgl, nullptr, qgb, NTOK, HDIM, HDIM, 0, QSCALE);

    attn_local_kernel<<<dim3(16, 4, NBATCH * NHEAD), 256, 0, stream>>>(
        qb, kb, vb, kgb, vgb, maskf, ctx);
    attn_global_kernel<<<dim3(8, 1, NBATCH * NHEAD), 256, 0, stream>>>(
        qgb, kgb, vgb, maskf, ctx);

    cvt(ctx, ctxb, (size_t)NTOK * HDIM);
    gemm(ctxb, wo, bol, attn, nullptr, NTOK, HDIM, HDIM, 0, 1.0f);
    add_ln_kernel<<<NTOK, 256, 0, stream>>>(xf, attn, ln1g + l * HDIM,
                                            ln1b + l * HDIM, xf, xb);

    gemm(xb,  w1, b1l, nullptr, f1b, NTOK, FFDIM, HDIM, 1, 1.0f);
    gemm(f1b, w2, b2l, f2, nullptr, NTOK, HDIM, FFDIM, 0, 1.0f);
    add_ln_kernel<<<NTOK, 256, 0, stream>>>(xf, f2, ln2g + l * HDIM,
                                            ln2b + l * HDIM, xf, xb);
  }

  head_kernel<<<1, 256, 0, stream>>>(xf, Wd, bd, Wh, bhh, out);
}
